// MyModel_87522843558514
// MI455X (gfx1250) — compile-verified, matrix operand order checked by reading, not run on hardware
//
#include <hip/hip_runtime.h>

// Problem constants (from reference): VOCAB=6, EMB=64, HID=64, B=256, T=2048.
#define VOCAB  6
#define EMB    64
#define HID    64
#define G4H    256   // 4*HID gate width
#define BATCH  256
#define MB     16    // batch rows per workgroup (WMMA M dimension)
#define TCHUNK 128   // token steps staged per async refill

typedef __attribute__((ext_vector_type(16))) _Float16 v16h;
typedef __attribute__((ext_vector_type(8)))  float    v8f;
typedef __attribute__((ext_vector_type(4)))  int      v4i;

union HFrag { v16h v; _Float16 h[16]; };   // A/B operand fragment (8 VGPRs)
union CFrag { v8f  v; float    f[8];  };   // C/D accumulator fragment (8 VGPRs)

#define GAS __attribute__((address_space(1)))
#define LAS __attribute__((address_space(3)))

#if __has_builtin(__builtin_amdgcn_global_load_async_to_lds_b128) && \
    __has_builtin(__builtin_amdgcn_s_wait_asynccnt)
#define HAVE_ASYNC_LDS 1
#else
#define HAVE_ASYNC_LDS 0
#endif

// gfx1250 has a hardware transcendental tanh (v_tanh_f32).
#if __has_builtin(__builtin_amdgcn_tanhf)
__device__ __forceinline__ float fast_tanh(float x) { return __builtin_amdgcn_tanhf(x); }
#else
__device__ __forceinline__ float fast_tanh(float x) { return tanhf(x); }
#endif
// sigmoid(x) = 0.5 + 0.5*tanh(x/2): one TRANS op + FMA, no divergent libm path.
__device__ __forceinline__ float sigf(float x) { return fmaf(0.5f, fast_tanh(0.5f * x), 0.5f); }

// ---------------------------------------------------------------------------
// Kernel 1: fold x@W + b into a [VOCAB, 4H] lookup table (fwd and bwd).
// blockIdx 0..5 -> forward vocab row, 6..11 -> backward vocab row.
// ---------------------------------------------------------------------------
__global__ void xw_precompute_kernel(const float* __restrict__ emb,
                                     const float* __restrict__ W_f, const float* __restrict__ b_f,
                                     const float* __restrict__ W_b, const float* __restrict__ b_b,
                                     float* __restrict__ xWf, float* __restrict__ xWb) {
  const int  v   = blockIdx.x % VOCAB;
  const bool bwd = blockIdx.x >= VOCAB;
  const int  n   = threadIdx.x;                 // 0..255 gate column
  const float* W  = bwd ? W_b : W_f;
  const float* bb = bwd ? b_b : b_f;
  float acc = bb[n];
  #pragma unroll 8
  for (int k = 0; k < EMB; ++k) acc += emb[v * EMB + k] * W[k * G4H + n];
  (bwd ? xWb : xWf)[v * G4H + n] = acc;
}

// ---------------------------------------------------------------------------
// Kernel 2: forward masked LSTM over T steps for a 16-row batch tile.
// 256 threads = 8 wave32s; each wave owns two 16-column N-tiles of the
// [16 x 256] gate matrix z_t = xW[token_t] + h_{t-1} @ U.
// h@U via v_wmma_f32_16x16x32_f16 (K=64 -> 2 chained WMMAs per tile), with
// tokens streamed through LDS by double-buffered async-to-LDS copies.
// ---------------------------------------------------------------------------
__global__ void __launch_bounds__(256, 1)
lstm_fwd_kernel(const int* __restrict__ tokens, const float* __restrict__ U,
                const float* __restrict__ xW, float* __restrict__ hfwd, int T) {
  __shared__ float    zbuf[MB][G4H];           // gate pre-activations (16 KB)
  __shared__ _Float16 hbuf[MB][HID];           // h_{t-1} f16 for A fragments (2 KB)
  __shared__ float    xWs[VOCAB][G4H];         // xW LUT (6 KB)
  __shared__ int      tokbuf[2][MB][TCHUNK];   // async-staged token ring (16 KB)

  const int tid    = threadIdx.x;
  const int lane   = tid & 31;
  const int wave   = tid >> 5;                 // 0..7
  const int b0     = blockIdx.x * MB;
  const int laneHi = (lane >> 4) & 1;          // lanes 16..31 take the +8 K / +8 M half
  const int lane15 = lane & 15;

  // ---- prologue: xW LUT to LDS, zero h0 ----------------------------------
  for (int idx = tid; idx < VOCAB * G4H; idx += 256)
    xWs[idx >> 8][idx & 255] = xW[idx];
  for (int idx = tid; idx < MB * HID; idx += 256)
    hbuf[idx >> 6][idx & 63] = (_Float16)0.0f;

  // ---- loop-invariant B fragments straight from global U (f32 -> f16) ----
  // 16-bit B 32x16 striping: lane<16 -> N=lane, K pairs {0..7,16..23};
  // lane>=16 -> same N, K pairs shifted by +8 (mirrors A layout).
  const int nt0 = wave * 2;
  HFrag bf[2][2];
  #pragma unroll
  for (int tile = 0; tile < 2; ++tile) {
    int ncol = (nt0 + tile) * 16 + lane15;
    #pragma unroll
    for (int kc = 0; kc < 2; ++kc) {
      #pragma unroll
      for (int v = 0; v < 8; ++v) {
        int k0 = kc * 32 + ((v & 4) ? 16 : 0) + (v & 3) * 2 + (laneHi ? 8 : 0);
        bf[tile][kc].h[2 * v]     = (_Float16)U[(k0)     * G4H + ncol];
        bf[tile][kc].h[2 * v + 1] = (_Float16)U[(k0 + 1) * G4H + ncol];
      }
    }
  }

  // ---- token staging: thread -> (row m = tid/16, 8 ints at col (tid%16)*8)
  const int trow = tid >> 4;
  const int tcol = (tid & 15) * 8;
  auto stage_tokens = [&](int buf, int tc) {
    const int* g = &tokens[(size_t)(b0 + trow) * T + tc + tcol];
    int*       l = &tokbuf[buf][trow][tcol];
#if HAVE_ASYNC_LDS
    __builtin_amdgcn_global_load_async_to_lds_b128((GAS v4i*)(g),     (LAS v4i*)(l),     0, 0);
    __builtin_amdgcn_global_load_async_to_lds_b128((GAS v4i*)(g + 4), (LAS v4i*)(l + 4), 0, 0);
#else
    #pragma unroll
    for (int j = 0; j < 8; ++j) l[j] = g[j];
#endif
  };

  stage_tokens(0, 0);                          // chunk 0 in flight

  // per-thread recurrent state: 4 fixed (m,n) cells (16*64 / 256 threads)
  float creg[4], hreg[4];
  #pragma unroll
  for (int j = 0; j < 4; ++j) { creg[j] = 0.0f; hreg[j] = 0.0f; }

  const int nchunk = T / TCHUNK;
  for (int c = 0; c < nchunk; ++c) {
    const int buf = c & 1;
#if HAVE_ASYNC_LDS
    __builtin_amdgcn_s_wait_asynccnt(0);       // chunk c landed in LDS
#endif
    __syncthreads();
    if (c + 1 < nchunk) {
      stage_tokens(buf ^ 1, (c + 1) * TCHUNK); // overlap next copy with compute
      if (c + 2 < nchunk)                      // and prefetch one further ahead
        __builtin_prefetch(&tokens[(size_t)(b0 + trow) * T + (c + 2) * TCHUNK + tcol], 0, 1);
    }

    for (int tt = 0; tt < TCHUNK; ++tt) {
      __syncthreads();                         // hbuf from prev step visible;
                                               // prev zbuf readers done

      // A fragments: h_{t-1} [16(M) x 64(K)] f16, K split into two 32-chunks.
      // 16-bit A 16x32 striping: lane<16 -> M=lane, K pairs {0..7,16..23};
      // lane>=16 -> same M, K pairs shifted by +8.
      HFrag af[2];
      #pragma unroll
      for (int kc = 0; kc < 2; ++kc) {
        #pragma unroll
        for (int v = 0; v < 8; ++v) {
          int k0 = kc * 32 + ((v & 4) ? 16 : 0) + (v & 3) * 2 + (laneHi ? 8 : 0);
          af[kc].h[2 * v]     = hbuf[lane15][k0];
          af[kc].h[2 * v + 1] = hbuf[lane15][k0 + 1];
        }
      }

      #pragma unroll
      for (int tile = 0; tile < 2; ++tile) {
        int ncol = (nt0 + tile) * 16 + lane15;
        CFrag cacc;
        #pragma unroll
        for (int r = 0; r < 8; ++r) cacc.f[r] = 0.0f;   // SRC2 = inline zero
        cacc.v = __builtin_amdgcn_wmma_f32_16x16x32_f16(false, af[0].v, false, bf[tile][0].v,
                                                        (short)0, cacc.v, false, false);
        cacc.v = __builtin_amdgcn_wmma_f32_16x16x32_f16(false, af[1].v, false, bf[tile][1].v,
                                                        (short)0, cacc.v, false, false);
        // C/D layout: VGPR r, lanes 0-15 -> M=r, lanes 16-31 -> M=r+8, N=lane&15.
        #pragma unroll
        for (int r = 0; r < 8; ++r)
          zbuf[r + (laneHi ? 8 : 0)][ncol] = cacc.f[r];
      }
      __syncthreads();                         // zbuf complete

      // gate math; thread owns cells p = tid + 256*j  (m = p/64, n = p%64)
      #pragma unroll
      for (int j = 0; j < 4; ++j) {
        int p = tid + 256 * j;
        int m = p >> 6, n = p & 63;
        int tok = tokbuf[buf][m][tt];
        const float* xwrow = xWs[tok];
        float iv = zbuf[m][n]       + xwrow[n];
        float fv = zbuf[m][n + 64]  + xwrow[n + 64];
        float gv = zbuf[m][n + 128] + xwrow[n + 128];
        float ov = zbuf[m][n + 192] + xwrow[n + 192];
        float cn = sigf(fv) * creg[j] + sigf(iv) * fast_tanh(gv);
        float hn = sigf(ov) * fast_tanh(cn);
        bool msk = (tok != 0);                 // Keras mask: carry state through
        creg[j] = msk ? cn : creg[j];
        hreg[j] = msk ? hn : hreg[j];
        hbuf[m][n] = (_Float16)hreg[j];        // feeds next step's A matrix
      }
    }
  }

  #pragma unroll
  for (int j = 0; j < 4; ++j) {
    int p = tid + 256 * j;
    int m = p >> 6, n = p & 63;
    hfwd[(size_t)(b0 + m) * HID + n] = hreg[j];
  }
}

// ---------------------------------------------------------------------------
// Kernel 3: the backward LSTM contributes only its step-0 output (seq[:,-1]
// of the un-reversed sequence) -> one masked LSTM step from zero state,
// fused with the decode head: out = [h_fwd, h_bwd1] @ W_d + b_d.
// ---------------------------------------------------------------------------
__global__ void finalize_kernel(const int* __restrict__ tokens, const float* __restrict__ xWb,
                                const float* __restrict__ hfwd, const float* __restrict__ W_d,
                                const float* __restrict__ b_d, float* __restrict__ out, int T) {
  int b = blockIdx.x * blockDim.x + threadIdx.x;
  if (b >= BATCH) return;
  int  tok = tokens[(size_t)b * T + (T - 1)];
  bool msk = tok != 0;
  float acc = b_d[0];
  #pragma unroll 4
  for (int n = 0; n < HID; ++n) {
    acc += hfwd[(size_t)b * HID + n] * W_d[n];
    if (msk) {
      float iv = xWb[tok * G4H + n];
      float gv = xWb[tok * G4H + n + 128];
      float ov = xWb[tok * G4H + n + 192];
      // c0 = 0 -> forget gate irrelevant on the first step
      float cn = sigf(iv) * fast_tanh(gv);
      float hn = sigf(ov) * fast_tanh(cn);
      acc += hn * W_d[HID + n];
    }
  }
  out[b] = acc;
}

// ---------------------------------------------------------------------------
extern "C" void kernel_launch(void* const* d_in, const int* in_sizes, int n_in,
                              void* d_out, int out_size, void* d_ws, size_t ws_size,
                              hipStream_t stream) {
  const int*   tokens = (const int*)  d_in[0];
  const float* emb    = (const float*)d_in[1];
  const float* W_f    = (const float*)d_in[2];
  const float* U_f    = (const float*)d_in[3];
  const float* b_f    = (const float*)d_in[4];
  const float* W_b    = (const float*)d_in[5];
  const float* U_b    = (const float*)d_in[6];   // unused: bwd needs only 1 step from zero state
  const float* b_b    = (const float*)d_in[7];
  const float* W_d    = (const float*)d_in[8];
  const float* b_d    = (const float*)d_in[9];
  (void)U_b; (void)n_in; (void)ws_size; (void)out_size;

  const int B = BATCH;
  const int T = in_sizes[0] / B;

  float* xWf  = (float*)d_ws;            // [VOCAB][4H]
  float* xWb  = xWf + VOCAB * G4H;       // [VOCAB][4H]
  float* hfwd = xWb + VOCAB * G4H;       // [B][HID]

  xw_precompute_kernel<<<2 * VOCAB, G4H, 0, stream>>>(emb, W_f, b_f, W_b, b_b, xWf, xWb);
  lstm_fwd_kernel<<<B / MB, 256, 0, stream>>>(tokens, U_f, xWf, hfwd, T);
  finalize_kernel<<<1, 256, 0, stream>>>(tokens, xWb, hfwd, W_d, b_d, (float*)d_out, T);
}
